// LSTMWithAttention_48584670052409
// MI455X (gfx1250) — matrix operand order repaired and checked by reading, not compiled, not run on hardware
//
#include <hip/hip_runtime.h>
#include <hip/hip_bf16.h>
#include <math.h>

// ---------------------------------------------------------------------------
// Problem constants (from the reference)
// ---------------------------------------------------------------------------
namespace cfg {
constexpr int kB    = 64;    // batch
constexpr int kT    = 20;    // seq len
constexpr int kDin  = 512;   // input dim
constexpr int kH    = 512;   // hidden
constexpr int kE    = 2048;  // encoder dim
constexpr int kP    = 196;   // pixels
constexpr int kA    = 512;   // attention dim
constexpr int kComb = kDin + kH + kE;  // 3072
}

typedef __attribute__((ext_vector_type(16))) __bf16 v16bf;
typedef __attribute__((ext_vector_type(8)))  __bf16 v8bf;
typedef __attribute__((ext_vector_type(8)))  float  v8f;

// ---------------------------------------------------------------------------
// B-matrix pre-swizzle: fp32 row-major W[K,N] -> bf16 fragment-ready layout.
//   Bp[ ((n_tile * (K/32) + kc) * 32 + lane) * 16 + e ]
//     = bf16( W[ kc*32 + (lane>>4)*16 + e ][ n_tile*16 + (lane&15) ] )
// so a wave's B fragment load is two coalesced global_load_b128 per lane.
// ---------------------------------------------------------------------------
__global__ __launch_bounds__(256)
void swizzle_b_kernel(const float* __restrict__ W, int ldw, int K,
                      __bf16* __restrict__ Bp, int total)
{
    const int idx = blockIdx.x * 256 + threadIdx.x;
    if (idx >= total) return;
    const int e     = idx & 15;
    const int lane  = (idx >> 4) & 31;
    const int rest  = idx >> 9;
    const int nkc   = K >> 5;
    const int kc    = rest % nkc;
    const int ntile = rest / nkc;
    const int n = ntile * 16 + (lane & 15);
    const int k = kc * 32 + (lane >> 4) * 16 + e;
    Bp[idx] = (__bf16)W[(size_t)k * ldw + n];
}

// ---------------------------------------------------------------------------
// GEMM: C[M,N] = A[M,K] (fp32 row-major) * Bp (pre-swizzled bf16) + bias[N].
// Block: 256 threads = 8 wave32 in a 2x4 grid; block tile 64x128, K-slab 64.
// Each wave owns a 32x32 tile (2x2 WMMA tiles): per 32-deep chunk it loads
// 2 A fragments (ds_load_b128 x4) + 2 B fragments (global b128 x4) and runs
// 4 x v_wmma_f32_16x16x32_bf16 -> 8 WMMA per slab per wave, 2 loads/WMMA.
// A staged into double-buffered LDS (fp32->bf16); B direct from L2-resident Bp.
// Requires M%64==0 (or M==64), N%128==0, K%64==0 — true for all call sites.
// ---------------------------------------------------------------------------
#define GEMM_BM 64
#define GEMM_BN 128
#define GEMM_BK 64
#define A_STRIDE 72   // padded bf16 row stride (144B = 9*16B: aligned, conflict-spread)

__global__ __launch_bounds__(256)
void gemm_bf16_wmma(const float* __restrict__ A, int lda,
                    const __bf16* __restrict__ Bp,
                    float* __restrict__ C, int ldc,
                    const float* __restrict__ bias,
                    int K)
{
    __shared__ alignas(16) __bf16 As[2][GEMM_BM][A_STRIDE];

    const int tid  = threadIdx.x;
    const int lane = tid & 31;
    const int wave = tid >> 5;
    const int wm   = wave >> 2;   // 32-row group (0..1)
    const int wn   = wave & 3;    // 32-col group (0..3)
    const int half = lane >> 4;
    const int l15  = lane & 15;

    const int block_m = blockIdx.x * GEMM_BM;
    const int block_n = blockIdx.y * GEMM_BN;

    const int nkc    = K >> 5;                    // # of 32-deep WMMA chunks
    const int ntiles = K >> 6;                    // # of 64-deep slabs
    const int t0     = blockIdx.y * 8 + wn * 2;   // wave's first global 16-col tile
    const int t1     = t0 + 1;

    // A staging coordinates: 16 consecutive fp32 per thread
    const int a_row = tid >> 2;             // 0..63
    const int a_col = (tid & 3) * 16;       // 0,16,32,48

    v8f acc00 = {}, acc01 = {}, acc10 = {}, acc11 = {};

    auto stage_store = [&](int buf, const float4& r0, const float4& r1,
                           const float4& r2, const float4& r3) {
        v8bf p0, p1;
        p0[0] = (__bf16)r0.x; p0[1] = (__bf16)r0.y; p0[2] = (__bf16)r0.z; p0[3] = (__bf16)r0.w;
        p0[4] = (__bf16)r1.x; p0[5] = (__bf16)r1.y; p0[6] = (__bf16)r1.z; p0[7] = (__bf16)r1.w;
        p1[0] = (__bf16)r2.x; p1[1] = (__bf16)r2.y; p1[2] = (__bf16)r2.z; p1[3] = (__bf16)r2.w;
        p1[4] = (__bf16)r3.x; p1[5] = (__bf16)r3.y; p1[6] = (__bf16)r3.z; p1[7] = (__bf16)r3.w;
        *(v8bf*)&As[buf][a_row][a_col]     = p0;   // ds_store_b128
        *(v8bf*)&As[buf][a_row][a_col + 8] = p1;   // ds_store_b128
    };

    // load an A fragment (16 rows x 32 K) from LDS: two aligned b128 chunks
    auto load_afrag = [&](int buf, int row_tile, int kc) -> v16bf {
        const __bf16* arow = &As[buf][row_tile * 16 + l15][kc * 32];
        v8bf lo = *(const v8bf*)(arow + half * 16);
        v8bf hi = *(const v8bf*)(arow + half * 16 + 8);
        return __builtin_shufflevector(lo, hi,
                0,1,2,3,4,5,6,7,8,9,10,11,12,13,14,15);
    };
    // load a B fragment (32 K x 16 cols) straight from pre-swizzled global
    auto load_bfrag = [&](int n_tile, int kcg) -> v16bf {
        const __bf16* bp = Bp + (((size_t)n_tile * nkc + kcg) * 32 + lane) * 16;
        return __builtin_shufflevector(*(const v8bf*)bp, *(const v8bf*)(bp + 8),
                0,1,2,3,4,5,6,7,8,9,10,11,12,13,14,15);
    };

    // ---- prologue: stage slab 0 ----------------------------------------
    {
        const float4* src = (const float4*)(A + (size_t)(block_m + a_row) * lda + a_col);
        stage_store(0, src[0], src[1], src[2], src[3]);
    }
    __syncthreads();

    for (int kt = 0; kt < ntiles; ++kt) {
        const int buf = kt & 1;
        const bool has_next = (kt + 1) < ntiles;

        // issue next slab's global loads before the WMMAs of this slab
        float4 r0, r1, r2, r3;
        if (has_next) {
            const float4* src = (const float4*)(A + (size_t)(block_m + a_row) * lda
                                                + (kt + 1) * GEMM_BK + a_col);
            r0 = src[0]; r1 = src[1]; r2 = src[2]; r3 = src[3];
        }
        if (kt + 2 < ntiles) {
            __builtin_prefetch(A + (size_t)(block_m + a_row) * lda
                               + (kt + 2) * GEMM_BK + a_col, 0, 3);
        }

        // ---- compute: two 32-deep chunks, 2x2 WMMA tiles each -----------
        #pragma unroll
        for (int kc = 0; kc < 2; ++kc) {
            const int kcg = kt * 2 + kc;
            v16bf af0 = load_afrag(buf, wm * 2 + 0, kc);
            v16bf af1 = load_afrag(buf, wm * 2 + 1, kc);
            v16bf bf0 = load_bfrag(t0, kcg);
            v16bf bf1 = load_bfrag(t1, kcg);

            acc00 = __builtin_amdgcn_wmma_f32_16x16x32_bf16(false, af0, false, bf0,
                                                            (short)0, acc00, false, false);
            acc01 = __builtin_amdgcn_wmma_f32_16x16x32_bf16(false, af0, false, bf1,
                                                            (short)0, acc01, false, false);
            acc10 = __builtin_amdgcn_wmma_f32_16x16x32_bf16(false, af1, false, bf0,
                                                            (short)0, acc10, false, false);
            acc11 = __builtin_amdgcn_wmma_f32_16x16x32_bf16(false, af1, false, bf1,
                                                            (short)0, acc11, false, false);
        }

        // store next slab into the other LDS buffer, then one barrier
        if (has_next) stage_store(buf ^ 1, r0, r1, r2, r3);
        __syncthreads();
    }

    // ---- epilogue (C/D layout: VGPR v -> row v + half*8; lanes 0-15 -> cols)
    {
        const int n0 = block_n + wn * 32 + l15;
        const int n1 = n0 + 16;
        const float bb0 = bias ? bias[n0] : 0.0f;
        const float bb1 = bias ? bias[n1] : 0.0f;
        const int m0 = block_m + wm * 32 + half * 8;
        const int m1 = m0 + 16;
        #pragma unroll
        for (int v = 0; v < 8; ++v) {
            C[(size_t)(m0 + v) * ldc + n0] = acc00[v] + bb0;
            C[(size_t)(m0 + v) * ldc + n1] = acc01[v] + bb1;
            C[(size_t)(m1 + v) * ldc + n0] = acc10[v] + bb0;
            C[(size_t)(m1 + v) * ldc + n1] = acc11[v] + bb1;
        }
    }
}

// ---------------------------------------------------------------------------
// Attention: per batch element b:
//   e[p] = sum_a v[a]*tanh(feat_proj[b,p,a]+hproj[b,a]);  alpha = softmax(e)
//   ctx[b] = sum_p alpha[p] * cnn[b,p,:]
// ---------------------------------------------------------------------------
__global__ __launch_bounds__(256)
void attn_kernel(const float* __restrict__ feat_proj,  // [B,P,A]
                 const float* __restrict__ hproj,      // [B,A]
                 const float* __restrict__ vvec,       // [A]
                 const float* __restrict__ cnn,        // [B,P,E]
                 float* __restrict__ ctx)              // [B,E]
{
    using namespace cfg;
    const int b    = blockIdx.x;
    const int tid  = threadIdx.x;
    const int lane = tid & 31;
    const int wave = tid >> 5;

    __shared__ float sE[kP];
    __shared__ float sRed[8];

    const float* hp = hproj + (size_t)b * kA;

    for (int p = wave; p < kP; p += 8) {
        const float* fp = feat_proj + ((size_t)b * kP + p) * kA;
        float sum = 0.0f;
        for (int a = lane; a < kA; a += 32)
            sum += vvec[a] * tanhf(fp[a] + hp[a]);
        #pragma unroll
        for (int off = 16; off > 0; off >>= 1)
            sum += __shfl_down(sum, off, 32);
        if (lane == 0) sE[p] = sum;
    }
    __syncthreads();

    float m = -INFINITY;
    for (int p = tid; p < kP; p += 256) m = fmaxf(m, sE[p]);
    #pragma unroll
    for (int off = 16; off > 0; off >>= 1) m = fmaxf(m, __shfl_down(m, off, 32));
    if (lane == 0) sRed[wave] = m;
    __syncthreads();
    if (tid == 0) {
        float mm = sRed[0];
        #pragma unroll
        for (int w = 1; w < 8; ++w) mm = fmaxf(mm, sRed[w]);
        sRed[0] = mm;
    }
    __syncthreads();
    const float gmax = sRed[0];
    __syncthreads();

    float s = 0.0f;
    for (int p = tid; p < kP; p += 256) {
        const float e = __expf(sE[p] - gmax);
        sE[p] = e;
        s += e;
    }
    #pragma unroll
    for (int off = 16; off > 0; off >>= 1) s += __shfl_down(s, off, 32);
    if (lane == 0) sRed[wave] = s;
    __syncthreads();
    if (tid == 0) {
        float ss = 0.0f;
        #pragma unroll
        for (int w = 0; w < 8; ++w) ss += sRed[w];
        sRed[0] = ss;
    }
    __syncthreads();
    const float inv = 1.0f / sRed[0];

    const float* cb = cnn + (size_t)b * kP * kE;
    for (int e0 = tid; e0 < kE; e0 += 256) {
        float acc = 0.0f;
        for (int p = 0; p < kP; ++p)
            acc += sE[p] * cb[(size_t)p * kE + e0];
        ctx[(size_t)b * kE + e0] = acc * inv;
    }
}

// ---------------------------------------------------------------------------
// comb[b,:] = [ x[b,t,:], h[b,:], ctx[b,:] ]
// ---------------------------------------------------------------------------
__global__ __launch_bounds__(256)
void build_comb_kernel(const float* __restrict__ x, int t,
                       const float* __restrict__ h,
                       const float* __restrict__ ctx,
                       float* __restrict__ comb)
{
    using namespace cfg;
    const int idx = blockIdx.x * 256 + threadIdx.x;
    if (idx >= kB * kComb) return;
    const int b   = idx / kComb;
    const int col = idx - b * kComb;
    float val;
    if (col < kDin)           val = x[((size_t)b * kT + t) * kDin + col];
    else if (col < kDin + kH) val = h[(size_t)b * kH + (col - kDin)];
    else                      val = ctx[(size_t)b * kE + (col - kDin - kH)];
    comb[idx] = val;
}

// ---------------------------------------------------------------------------
// Pointwise LSTM update. gates[b, 0:H)=i, [H:2H)=f, [2H:3H)=c~, [3H:4H)=o
// ---------------------------------------------------------------------------
__device__ __forceinline__ float sigmoidf_(float x) { return 1.0f / (1.0f + __expf(-x)); }

__global__ __launch_bounds__(256)
void lstm_pointwise_kernel(const float* __restrict__ gates,
                           float* __restrict__ h,
                           float* __restrict__ c,
                           float* __restrict__ out,     // [B,T,H]
                           float* __restrict__ hT,      // [B,H]
                           float* __restrict__ cT,      // [B,H]
                           int t, int last)
{
    using namespace cfg;
    const int idx = blockIdx.x * 256 + threadIdx.x;
    if (idx >= kB * kH) return;
    const int b = idx / kH;
    const int j = idx - b * kH;
    const float* g = gates + (size_t)b * 4 * kH;
    const float it = sigmoidf_(g[j]);
    const float ft = sigmoidf_(g[kH + j]);
    const float cc = tanhf(g[2 * kH + j]);
    const float ot = sigmoidf_(g[3 * kH + j]);
    const float cn = ft * c[idx] + it * cc;
    const float hn = ot * tanhf(cn);
    c[idx] = cn;
    h[idx] = hn;
    out[((size_t)b * kT + t) * kH + j] = hn;
    if (last) { hT[idx] = hn; cT[idx] = cn; }
}

__global__ __launch_bounds__(256)
void zero_kernel(float* __restrict__ p, int n)
{
    const int idx = blockIdx.x * 256 + threadIdx.x;
    if (idx < n) p[idx] = 0.0f;
}

// ---------------------------------------------------------------------------
// Host-side orchestration
// ---------------------------------------------------------------------------
extern "C" void kernel_launch(void* const* d_in, const int* in_sizes, int n_in,
                              void* d_out, int out_size, void* d_ws, size_t ws_size,
                              hipStream_t stream)
{
    using namespace cfg;
    (void)in_sizes; (void)n_in; (void)out_size; (void)ws_size;

    // inputs in setup_inputs() order
    const float* x       = (const float*)d_in[0];   // [B,T,D_IN]
    const float* cnn     = (const float*)d_in[1];   // [B,P,E]
    const float* Wi      = (const float*)d_in[2];   // [3072,512]
    const float* bi      = (const float*)d_in[3];
    const float* Wf      = (const float*)d_in[4];
    const float* bf      = (const float*)d_in[5];
    const float* Wc      = (const float*)d_in[6];
    const float* bc      = (const float*)d_in[7];
    const float* Wo      = (const float*)d_in[8];
    const float* bo      = (const float*)d_in[9];
    const float* Wa_feat = (const float*)d_in[10];  // [E,A]
    const float* Wa_h    = (const float*)d_in[11];  // [H,A]
    const float* ba      = (const float*)d_in[12];  // [A]
    const float* vvec    = (const float*)d_in[13];  // [A]

    float* out = (float*)d_out;                     // [B,T,H] ++ [B,H] ++ [B,H]
    float* hT  = out + (size_t)kB * kT * kH;
    float* cT  = hT + (size_t)kB * kH;

    // workspace carve-up, 256B aligned
    auto align256 = [](size_t v) { return (v + 255) & ~(size_t)255; };
    char* ws = (char*)d_ws;
    size_t off = 0;
    float* feat_proj = (float*)(ws + off); off = align256(off + (size_t)kB * kP * kA * 4);
    float* hproj     = (float*)(ws + off); off = align256(off + (size_t)kB * kA * 4);
    float* ctx       = (float*)(ws + off); off = align256(off + (size_t)kB * kE * 4);
    float* comb      = (float*)(ws + off); off = align256(off + (size_t)kB * kComb * 4);
    float* gates     = (float*)(ws + off); off = align256(off + (size_t)kB * 4 * kH * 4);
    float* h         = (float*)(ws + off); off = align256(off + (size_t)kB * kH * 4);
    float* c         = (float*)(ws + off); off = align256(off + (size_t)kB * kH * 4);
    // pre-swizzled bf16 weights
    __bf16* WaFp = (__bf16*)(ws + off); off = align256(off + (size_t)kE * kA * 2);
    __bf16* WaHp = (__bf16*)(ws + off); off = align256(off + (size_t)kH * kA * 2);
    __bf16* Wgp[4];
    Wgp[0] = (__bf16*)(ws + off); off = align256(off + (size_t)kComb * kH * 2);
    Wgp[1] = (__bf16*)(ws + off); off = align256(off + (size_t)kComb * kH * 2);
    Wgp[2] = (__bf16*)(ws + off); off = align256(off + (size_t)kComb * kH * 2);
    Wgp[3] = (__bf16*)(ws + off); off = align256(off + (size_t)kComb * kH * 2);

    // ---- pre-swizzle weights into fragment-ready bf16 layout ----
    {
        int tot = kE * kA;
        swizzle_b_kernel<<<(tot + 255) / 256, 256, 0, stream>>>(Wa_feat, kA, kE, WaFp, tot);
        tot = kH * kA;
        swizzle_b_kernel<<<(tot + 255) / 256, 256, 0, stream>>>(Wa_h, kA, kH, WaHp, tot);
        tot = kComb * kH;
        swizzle_b_kernel<<<(tot + 255) / 256, 256, 0, stream>>>(Wi, kH, kComb, Wgp[0], tot);
        swizzle_b_kernel<<<(tot + 255) / 256, 256, 0, stream>>>(Wf, kH, kComb, Wgp[1], tot);
        swizzle_b_kernel<<<(tot + 255) / 256, 256, 0, stream>>>(Wc, kH, kComb, Wgp[2], tot);
        swizzle_b_kernel<<<(tot + 255) / 256, 256, 0, stream>>>(Wo, kH, kComb, Wgp[3], tot);
    }

    // h0 = c0 = 0
    {
        const int n = kB * kH;
        zero_kernel<<<(n + 255) / 256, 256, 0, stream>>>(h, n);
        zero_kernel<<<(n + 255) / 256, 256, 0, stream>>>(c, n);
    }

    // feat_proj = cnn @ Wa_feat + ba   ([B*P, E] x [E, A])
    {
        dim3 grid((kB * kP) / GEMM_BM, kA / GEMM_BN);
        gemm_bf16_wmma<<<grid, 256, 0, stream>>>(cnn, kE, WaFp, feat_proj, kA, ba, kE);
    }

    const float* gate_bias[4] = { bi, bf, bc, bo };

    for (int t = 0; t < kT; ++t) {
        // hproj = h @ Wa_h
        {
            dim3 grid(kB / GEMM_BM, kA / GEMM_BN);
            gemm_bf16_wmma<<<grid, 256, 0, stream>>>(h, kH, WaHp, hproj, kA, nullptr, kH);
        }
        // attention scores + softmax + context
        attn_kernel<<<kB, 256, 0, stream>>>(feat_proj, hproj, vvec, cnn, ctx);

        // comb = [x_t, h, ctx]
        {
            const int n = kB * kComb;
            build_comb_kernel<<<(n + 255) / 256, 256, 0, stream>>>(x, t, h, ctx, comb);
        }

        // 4 gate GEMMs: comb @ W* + b*  -> gates[:, g*512 : (g+1)*512]
        {
            dim3 grid(kB / GEMM_BM, kH / GEMM_BN);
            #pragma unroll
            for (int g = 0; g < 4; ++g)
                gemm_bf16_wmma<<<grid, 256, 0, stream>>>(comb, kComb, Wgp[g],
                                                         gates + g * kH, 4 * kH,
                                                         gate_bias[g], kComb);
        }

        // pointwise LSTM update
        {
            const int n = kB * kH;
            lstm_pointwise_kernel<<<(n + 255) / 256, 256, 0, stream>>>(
                gates, h, c, out, hT, cT, t, t == kT - 1 ? 1 : 0);
        }
    }
}
